// DeepVtx_5609227289124
// MI455X (gfx1250) — matrix operand order repaired and checked by reading, not compiled, not run on hardware
//
#include <hip/hip_runtime.h>
#include <stdint.h>

// ---------------------------------------------------------------------------
// Types for CDNA5 WMMA (wave32): v_wmma_f32_16x16x32_bf16
// ---------------------------------------------------------------------------
typedef __attribute__((ext_vector_type(16))) __bf16          v16bf;
typedef __attribute__((ext_vector_type(16))) unsigned short  v16us;
typedef __attribute__((ext_vector_type(8)))  float           v8f;

#define CONV_WAVES 4
#define KMAX 256      // max Cin across the net (bottleneck / concat layers)
#define APAD 264      // padded A-row length (ushorts): 528B rows, 16B aligned, breaks bank conflicts

__device__ __forceinline__ unsigned short f2bf(float f) {
  union { float f; uint32_t u; } v; v.f = f;
  uint32_t r = v.u + 0x7FFFu + ((v.u >> 16) & 1u);   // round-to-nearest-even
  return (unsigned short)(r >> 16);
}
__device__ __forceinline__ float bf2f(unsigned short h) {
  union { uint32_t u; float f; } v; v.u = ((uint32_t)h) << 16;
  return v.f;
}
__device__ __forceinline__ void atomAddF(float* p, float v) {
  // lowers to global_atomic_add_f32 (relaxed, device scope)
  __hip_atomic_fetch_add(p, v, __ATOMIC_RELAXED, __HIP_MEMORY_SCOPE_AGENT);
}
// generic (flat) shared pointer -> raw LDS byte offset (flat addr low 32 bits)
__device__ __forceinline__ unsigned ldsAddr(const void* p) {
  return (unsigned)(size_t)p;
}
// async 16-byte global -> LDS copy (CDNA5, ASYNCcnt-tracked), saddr+voffset form
__device__ __forceinline__ void asyncLdsB128(unsigned lds, const void* base, unsigned off) {
  asm volatile("global_load_async_to_lds_b128 %0, %1, %2"
               :: "v"(lds), "v"(off), "s"(base) : "memory");
}
__device__ __forceinline__ void waitAsync0() {
  asm volatile("s_wait_asynccnt 0" ::: "memory");
}

// ---------------------------------------------------------------------------
// Sparse conv: rulebook gather -> per-tap 16x16x32 bf16 WMMA -> scatter-add.
// One wave = one 16-row x 16-Cout output tile for tap blockIdx.z.
// A rows / B tile staged with global_load_async_to_lds_b128 (ASYNCcnt path);
// dummy rulebook entries and K-padding are zero-filled with LDS stores.
// ---------------------------------------------------------------------------
__global__ void __launch_bounds__(32 * CONV_WAVES)
DeepVtx_sparse_conv_wmma(const unsigned short* __restrict__ act,   // [n_in][Cin] bf16
                         const unsigned short* __restrict__ Wq,    // [K][Cin][Cout] bf16
                         const int* __restrict__ rb_in,            // [K][P]
                         const int* __restrict__ rb_out,           // [K][P]
                         float* __restrict__ out,                  // [n_out][Cout] f32 (pre-zeroed)
                         int Cin, int Cout, int P, int n_in, int n_out)
{
  __shared__ unsigned short sA[CONV_WAVES][16][APAD];
  __shared__ unsigned short sB[KMAX][16];

  const int tid   = threadIdx.x;
  const int wave  = tid >> 5;
  const int lane  = tid & 31;
  const int half  = lane >> 4;
  const int l16   = lane & 15;
  const int k     = blockIdx.z;
  const int cout0 = blockIdx.y * 16;
  const int p0    = (blockIdx.x * CONV_WAVES + wave) * 16;
  const int KP    = (Cin + 31) & ~31;        // K padded to WMMA depth

  // ---- stage B tile: W[k][0:KP][cout0:cout0+16]; async 16B segments, zero pad ----
  for (int i = tid; i < KP * 2; i += 32 * CONV_WAVES) {
    int kk = i >> 1, seg = i & 1;            // 8-ushort (16B) halves of a 16-wide row
    unsigned lds = ldsAddr(&sB[kk][seg * 8]);
    if (kk < Cin) {
      unsigned off = (unsigned)((((size_t)k * Cin + kk) * Cout + cout0 + seg * 8) * 2);
      asyncLdsB128(lds, Wq, off);
    } else {
      *(uint4*)&sB[kk][seg * 8] = make_uint4(0u, 0u, 0u, 0u);
    }
  }

  // ---- stage A tile: 16 gathered activation rows ----
  {
    int pn = p0 + 16 * CONV_WAVES;
    if (pn < P) __builtin_prefetch(&rb_in[(size_t)k * P + pn], 0, 0); // global_prefetch_b8
  }
  if (Cin >= 8) {
    // async path: 16B (8 bf16) segments per row; Cin is a multiple of 8 here
    const int segs = KP >> 3;
    for (int i = lane; i < 16 * segs; i += 32) {
      int r  = i / segs, sg = i % segs;
      int p  = p0 + r;
      int src = (p < P) ? rb_in[(size_t)k * P + p] : n_in;
      unsigned lds = ldsAddr(&sA[wave][r][sg * 8]);
      if ((src >= 0) && (src < n_in) && (sg * 8 < Cin)) {
        unsigned off = (unsigned)(((size_t)src * Cin + sg * 8) * 2);
        asyncLdsB128(lds, act, off);
      } else {
        *(uint4*)&sA[wave][r][sg * 8] = make_uint4(0u, 0u, 0u, 0u);
      }
    }
  } else {
    // stem (Cin == 3): scalar gather
    for (int r = 0; r < 16; ++r) {
      int  p   = p0 + r;
      int  src = (p < P) ? rb_in[(size_t)k * P + p] : n_in;
      bool ok  = (src >= 0) && (src < n_in);
      for (int c = lane; c < KP; c += 32)
        sA[wave][r][c] = (ok && c < Cin) ? act[(size_t)src * Cin + c]
                                         : (unsigned short)0;
    }
  }
  waitAsync0();          // this wave's async LDS writes complete
  __syncthreads();       // cross-wave visibility (B tile shared by block)

  // ---- accumulate: D = sum_kb A(16x32) x B(32x16), f32 accumulators ----
  v8f acc = {};
  for (int kb = 0; kb < KP; kb += 32) {
    v16us au, bu;
#pragma unroll
    for (int j = 0; j < 16; ++j) {
      // A layout (16-bit A 16x32, ISA 7.12.2): lane half selects K sub-bands
      int ka = kb + j + 8 * half + ((j >= 8) ? 8 : 0);
      au[j]  = sA[wave][l16][ka];
      // B layout (32x16): same K order as A (lane half selects K sub-bands)
      int kbb = kb + j + 8 * half + ((j >= 8) ? 8 : 0);
      bu[j]   = sB[kbb][l16];
    }
    v16bf a = __builtin_bit_cast(v16bf, au);
    v16bf b = __builtin_bit_cast(v16bf, bu);
    acc = __builtin_amdgcn_wmma_f32_16x16x32_bf16(false, a, false, b,
                                                  (short)0, acc, false, false);
  }

  // ---- scatter-add C tile (C layout: VGPR r -> M = r + 8*half, N = lane&15) ----
#pragma unroll
  for (int r = 0; r < 8; ++r) {
    int M = r + 8 * half;
    int p = p0 + M;
    if (p < P) {
      int o = rb_out[(size_t)k * P + p];
      if (o >= 0 && o < n_out)
        atomAddF(&out[(size_t)o * Cout + cout0 + l16], acc[r]);
    }
  }
}

// ---------------------------------------------------------------------------
// Support kernels
// ---------------------------------------------------------------------------
__global__ void DeepVtx_zero_f32(float* p, long n) {
  long i = (long)blockIdx.x * 256 + threadIdx.x;
  if (i < n) p[i] = 0.f;
}

__global__ void DeepVtx_f32_to_bf16(const float* __restrict__ x,
                                    unsigned short* __restrict__ y, long n) {
  long i = (long)blockIdx.x * 256 + threadIdx.x;
  if (i < n) y[i] = f2bf(x[i]);
}

// per-channel sum / sumsq via device-scope f32 atomics (stats pre-zeroed)
__global__ void DeepVtx_bn_stats(const float* __restrict__ x,
                                 float* __restrict__ stats, int n, int C) {
  int tid = blockIdx.x * blockDim.x + threadIdx.x;
  int c = tid % C;
  int r = tid / C;
  int rstride = (gridDim.x * blockDim.x) / C;
  float s = 0.f, s2 = 0.f;
  for (; r < n; r += rstride) {
    float v = x[(size_t)r * C + c];
    s += v; s2 += v * v;
  }
  atomAddF(&stats[c], s);
  atomAddF(&stats[C + c], s2);
}

// fused BN + ReLU + bf16 quantize (activations feed the WMMA conv)
__global__ void DeepVtx_bn_apply(const float* __restrict__ x,
                                 const float* __restrict__ stats,
                                 const float* __restrict__ g,
                                 const float* __restrict__ b,
                                 unsigned short* __restrict__ y, int n, int C) {
  long i = (long)blockIdx.x * 256 + threadIdx.x;
  if (i >= (long)n * C) return;
  int c = (int)(i % C);
  float inv_n = 1.f / (float)n;
  float mu  = stats[c] * inv_n;
  float var = stats[C + c] * inv_n - mu * mu;
  float sc  = g[c] * rsqrtf(var + 1e-4f);
  float sh  = b[c] - mu * sc;
  float v   = x[i] * sc + sh;
  y[i] = f2bf(v > 0.f ? v : 0.f);
}

// scn.JoinTable: o[:, :A] = a, o[:, A:] = y
__global__ void DeepVtx_concat(const float* __restrict__ a,
                               const float* __restrict__ y,
                               float* __restrict__ o, int n, int A) {
  long i = (long)blockIdx.x * 256 + threadIdx.x;
  if (i >= (long)n * 2 * A) return;
  int r = (int)(i / (2 * A)), c = (int)(i % (2 * A));
  o[i] = (c < A) ? a[(size_t)r * A + c] : y[(size_t)r * A + (c - A)];
}

__global__ void DeepVtx_linear_sigmoid(const unsigned short* __restrict__ x,  // [n][16] bf16
                                       const float* __restrict__ w,           // [16][2]
                                       const float* __restrict__ bias,        // [2]
                                       float* __restrict__ out, int n) {
  int r = blockIdx.x * 256 + threadIdx.x;
  if (r >= n) return;
  float a0 = bias[0], a1 = bias[1];
#pragma unroll
  for (int c = 0; c < 16; ++c) {
    float v = bf2f(x[r * 16 + c]);
    a0 += v * w[2 * c];
    a1 += v * w[2 * c + 1];
  }
  out[2 * r]     = 1.f / (1.f + __expf(-a0));
  out[2 * r + 1] = 1.f / (1.f + __expf(-a1));
}

// ---------------------------------------------------------------------------
// Host orchestration
// ---------------------------------------------------------------------------
extern "C" void kernel_launch(void* const* d_in, const int* in_sizes, int n_in_args,
                              void* d_out, int out_size, void* d_ws, size_t ws_size,
                              hipStream_t stream) {
  (void)n_in_args; (void)out_size; (void)ws_size;
  static const int NP[5] = {16, 32, 64, 128, 256};

  // ---- input walk: setup_inputs insertion order, params flattened depth-first
  int ix = 0;
  const float* feats  = (const float*)d_in[ix++];
  const float* w_init = (const float*)d_in[ix++];
  struct LvlP {
    const float *gp1,*bp1,*wp1,*gp2,*bp2,*wp2,*gd,*bd,*wd,
                *gu,*bu,*wu,*gq1,*bq1,*wq1,*gq2,*bq2,*wq2;
  } LP[4];
  for (int l = 0; l < 4; ++l) {
    LP[l].gp1=(const float*)d_in[ix++]; LP[l].bp1=(const float*)d_in[ix++]; LP[l].wp1=(const float*)d_in[ix++];
    LP[l].gp2=(const float*)d_in[ix++]; LP[l].bp2=(const float*)d_in[ix++]; LP[l].wp2=(const float*)d_in[ix++];
    LP[l].gd =(const float*)d_in[ix++]; LP[l].bd =(const float*)d_in[ix++]; LP[l].wd =(const float*)d_in[ix++];
    LP[l].gu =(const float*)d_in[ix++]; LP[l].bu =(const float*)d_in[ix++]; LP[l].wu =(const float*)d_in[ix++];
    LP[l].gq1=(const float*)d_in[ix++]; LP[l].bq1=(const float*)d_in[ix++]; LP[l].wq1=(const float*)d_in[ix++];
    LP[l].gq2=(const float*)d_in[ix++]; LP[l].bq2=(const float*)d_in[ix++]; LP[l].wq2=(const float*)d_in[ix++];
  }
  const float* gB1=(const float*)d_in[ix++]; const float* bB1=(const float*)d_in[ix++];
  const float* wB1=(const float*)d_in[ix++];
  const float* gB2=(const float*)d_in[ix++]; const float* bB2=(const float*)d_in[ix++];
  const float* wB2=(const float*)d_in[ix++];
  const float* gF =(const float*)d_in[ix++]; const float* bF =(const float*)d_in[ix++];
  const float* linW=(const float*)d_in[ix++]; const float* linB=(const float*)d_in[ix++];

  const int* rsi[5]; const int* rso[5]; const int* rdi[4]; const int* rdo[4];
  int n[5], Pd[4];
  for (int l = 0; l < 5; ++l) { rsi[l]=(const int*)d_in[ix]; n[l]=in_sizes[ix]/27; ix++; }
  for (int l = 0; l < 5; ++l) { rso[l]=(const int*)d_in[ix]; ix++; }
  for (int l = 0; l < 4; ++l) { rdi[l]=(const int*)d_in[ix]; Pd[l]=in_sizes[ix]/8; ix++; }
  for (int l = 0; l < 4; ++l) { rdo[l]=(const int*)d_in[ix]; ix++; }
  // n_sites device scalars unused (derived from rulebook sizes above)

  // ---- workspace bump allocator
  char* wsp = (char*)d_ws; size_t wof = 0;
  auto alloc = [&](size_t bytes) -> void* {
    void* p = wsp + wof;
    wof = (wof + bytes + 255) & ~(size_t)255;
    return p;
  };

  auto cvt = [&](const float* src, long cnt) -> unsigned short* {
    unsigned short* dst = (unsigned short*)alloc((size_t)cnt * 2);
    DeepVtx_f32_to_bf16<<<(unsigned)((cnt + 255) / 256), 256, 0, stream>>>(src, dst, cnt);
    return dst;
  };

  // bf16 weight arenas (re-converted every call: deterministic, capture-safe)
  unsigned short* wqInit = cvt(w_init, 27L * 3 * 16);
  unsigned short *wqP1[4], *wqP2[4], *wqD[4], *wqU[4], *wqQ1[4], *wqQ2[4];
  for (int l = 0; l < 4; ++l) {
    long a = NP[l], bch = NP[l + 1];
    wqP1[l] = cvt(LP[l].wp1, 27L * a * a);
    wqP2[l] = cvt(LP[l].wp2, 27L * a * a);
    wqD [l] = cvt(LP[l].wd,   8L * a * bch);
    wqU [l] = cvt(LP[l].wu,   8L * bch * a);
    wqQ1[l] = cvt(LP[l].wq1, 27L * 2 * a * a);
    wqQ2[l] = cvt(LP[l].wq2, 27L * a * a);
  }
  unsigned short* wqB1 = cvt(wB1, 27L * 256 * 256);
  unsigned short* wqB2 = cvt(wB2, 27L * 256 * 256);

  // activation buffers: worst single-tensor element count across the net
  size_t maxE = (size_t)n[0] * 32;
  for (int l = 1; l < 4; ++l) { size_t e = (size_t)n[l] * 2 * NP[l]; if (e > maxE) maxE = e; }
  { size_t e = (size_t)n[4] * 256; if (e > maxE) maxE = e; }

  unsigned short* abf = (unsigned short*)alloc(maxE * 2);      // bf16 activation scratch
  float* stats = (float*)alloc(2 * 256 * sizeof(float));       // BN sum / sumsq
  float* pool[4]; int ps = 0;
  for (int i = 0; i < 4; ++i) pool[i] = (float*)alloc(maxE * sizeof(float));
  auto nslot = [&]() -> float* { float* p = pool[ps]; ps = (ps + 1) & 3; return p; };
  float* skipB[4];
  for (int l = 0; l < 4; ++l) skipB[l] = (float*)alloc((size_t)n[l] * NP[l] * sizeof(float));

  // ---- launch helpers
  auto zero = [&](float* p, long cnt) {
    DeepVtx_zero_f32<<<(unsigned)((cnt + 255) / 256), 256, 0, stream>>>(p, cnt);
  };
  auto conv = [&](const unsigned short* actb, const unsigned short* Wq,
                  const int* rin, const int* rout, float* outp,
                  int Cin, int Cout, int K, int P, int ni, int no) {
    zero(outp, (long)no * Cout);
    dim3 gr((unsigned)((P + 16 * CONV_WAVES - 1) / (16 * CONV_WAVES)),
            (unsigned)(Cout / 16), (unsigned)K);
    DeepVtx_sparse_conv_wmma<<<gr, 32 * CONV_WAVES, 0, stream>>>(
        actb, Wq, rin, rout, outp, Cin, Cout, P, ni, no);
  };
  auto bnrelu = [&](const float* xin, const float* g, const float* b, int nn, int C) {
    zero(stats, 2L * C);
    DeepVtx_bn_stats<<<64, 256, 0, stream>>>(xin, stats, nn, C);
    long tot = (long)nn * C;
    DeepVtx_bn_apply<<<(unsigned)((tot + 255) / 256), 256, 0, stream>>>(
        xin, stats, g, b, abf, nn, C);
  };

  // ---- network ----
  // stem: feats -> bf16 -> subm conv (3 -> 16)
  DeepVtx_f32_to_bf16<<<(unsigned)(((long)n[0] * 3 + 255) / 256), 256, 0, stream>>>(
      feats, abf, (long)n[0] * 3);
  float* x = nslot();
  conv(abf, wqInit, rsi[0], rso[0], x, 3, 16, 27, n[0], n[0], n[0]);

  // encoder
  for (int l = 0; l < 4; ++l) {
    int a = NP[l], bch = NP[l + 1];
    bnrelu(x, LP[l].gp1, LP[l].bp1, n[l], a);
    float* x1 = nslot();
    conv(abf, wqP1[l], rsi[l], rso[l], x1, a, a, 27, n[l], n[l], n[l]);
    bnrelu(x1, LP[l].gp2, LP[l].bp2, n[l], a);
    conv(abf, wqP2[l], rsi[l], rso[l], skipB[l], a, a, 27, n[l], n[l], n[l]);
    bnrelu(skipB[l], LP[l].gd, LP[l].bd, n[l], a);
    float* y = nslot();
    conv(abf, wqD[l], rdi[l], rdo[l], y, a, bch, 8, Pd[l], n[l], n[l + 1]);
    x = y;
  }

  // bottleneck
  bnrelu(x, gB1, bB1, n[4], 256);
  float* xb = nslot();
  conv(abf, wqB1, rsi[4], rso[4], xb, 256, 256, 27, n[4], n[4], n[4]);
  bnrelu(xb, gB2, bB2, n[4], 256);
  x = nslot();
  conv(abf, wqB2, rsi[4], rso[4], x, 256, 256, 27, n[4], n[4], n[4]);

  // decoder (deconv = transposed rulebook) + skip concat
  for (int l = 3; l >= 0; --l) {
    int a = NP[l], bch = NP[l + 1];
    bnrelu(x, LP[l].gu, LP[l].bu, n[l + 1], bch);
    float* y = nslot();
    conv(abf, wqU[l], rdo[l], rdi[l], y, bch, a, 8, Pd[l], n[l + 1], n[l]);
    float* cat = nslot();
    DeepVtx_concat<<<(unsigned)(((long)n[l] * 2 * a + 255) / 256), 256, 0, stream>>>(
        skipB[l], y, cat, n[l], a);
    bnrelu(cat, LP[l].gq1, LP[l].bq1, n[l], 2 * a);
    float* xp = nslot();
    conv(abf, wqQ1[l], rsi[l], rso[l], xp, 2 * a, a, 27, n[l], n[l], n[l]);
    bnrelu(xp, LP[l].gq2, LP[l].bq2, n[l], a);
    x = nslot();
    conv(abf, wqQ2[l], rsi[l], rso[l], x, a, a, 27, n[l], n[l], n[l]);
  }

  // head
  bnrelu(x, gF, bF, n[0], 16);
  DeepVtx_linear_sigmoid<<<(unsigned)((n[0] + 255) / 256), 256, 0, stream>>>(
      abf, linW, linB, (float*)d_out, n[0]);
}
